// ExplicitAttention_1520418422888
// MI455X (gfx1250) — matrix operand order repaired and checked by reading, not compiled, not run on hardware
//
#include <hip/hip_runtime.h>

// ---------------------------------------------------------------------------
// Flash attention with segment masking for MI455X (gfx1250, wave32, WMMA).
// B=1, H=16, S=4096, D=128. fp32 in/out, bf16 WMMA math with fp32 accum.
// BN=64 KV tiles: 32 v_wmma_f32_16x16x32_bf16 per iteration per wave.
// ---------------------------------------------------------------------------

#define H_   16
#define S_   4096
#define D_   128
#define BM   128   // q rows per block (8 waves x 16)
#define BN   64    // kv rows per iteration
#define NWAVE 8

typedef __attribute__((ext_vector_type(16))) __bf16 v16bf;
typedef __attribute__((ext_vector_type(8)))  float  v8f;

union ARegs { v16bf v; unsigned u[8]; };   // 32B: WMMA A/B operand
union CRegs { v8f   v; float    f[8]; };   // 32B: WMMA C/D accumulator

// Native bf16 convert (lowers to v_cvt_(pk_)bf16_f32, RNE).
__device__ __forceinline__ unsigned short bfbits(float x) {
    __bf16 b = (__bf16)x;
    unsigned short u;
    __builtin_memcpy(&u, &b, 2);
    return u;
}
__device__ __forceinline__ unsigned pack2(float lo, float hi) {
    return (unsigned)bfbits(lo) | ((unsigned)bfbits(hi) << 16);
}

__global__ __launch_bounds__(256, 1)
void flash_attn_seg_kernel(const float* __restrict__ q,
                           const float* __restrict__ k,
                           const float* __restrict__ v,
                           const int*   __restrict__ qseg,
                           const int*   __restrict__ kvseg,
                           float*       __restrict__ out)
{
    // ---- LDS (~83 KB total; WGP has 320 KB) ----
    __shared__ __align__(16) unsigned short Kld[BN][D_];        // K tile bf16 (16 KB)
    __shared__ __align__(16) unsigned short Vt[D_][BN];         // V^T   bf16 (16 KB)
    __shared__ int            ksegS[BN];
    __shared__ __align__(16) float          Sld[NWAVE][16][BN]; // scores f32 (32 KB)
    __shared__ __align__(16) unsigned short Pld[NWAVE][16][BN]; // P bf16     (16 KB)
    __shared__ float          rowAlpha[NWAVE][16];
    __shared__ float          rowStat[NWAVE][16];

    const int head   = blockIdx.x / (S_ / BM);
    const int qTile  = blockIdx.x % (S_ / BM);
    const int qBase  = qTile * BM;
    const int tid    = threadIdx.x;
    const int wave   = tid >> 5;
    const int lane   = tid & 31;
    const int laneLo = lane & 15;
    const int half   = lane >> 4;
    const int qWave  = qBase + wave * 16;
    const float scale = 0.08838834764831845f;     // 1/sqrt(128)

    // ---- preload Q strip into WMMA A-layout registers (pre-scaled bf16) ----
    // A 16x32 bf16, k order chosen here: lane holds row m = lane%16; VGPR j (0..3): K = 16*half+2j,+1;
    //               VGPR 4+j: K = 16*half+8+2j,+1.
    ARegs QA[4];
    {
        const float* qrow = q + ((size_t)head * S_ + (qWave + laneLo)) * D_;
#pragma unroll
        for (int kc = 0; kc < 4; ++kc) {
            const int dBase = kc * 32;
#pragma unroll
            for (int j = 0; j < 4; ++j) {
                const int d0 = dBase + 16 * half + 2 * j;
                const int d1 = dBase + 16 * half + 8 + 2 * j;
                QA[kc].u[j]     = pack2(qrow[d0] * scale, qrow[d0 + 1] * scale);
                QA[kc].u[4 + j] = pack2(qrow[d1] * scale, qrow[d1 + 1] * scale);
            }
        }
    }

    // q segment ids for the 8 C-layout rows this lane sees (m = vv + 8*half)
    int qs8[8];
#pragma unroll
    for (int vv = 0; vv < 8; ++vv) qs8[vv] = qseg[qWave + 8 * half + vv];

    // ---- O accumulator: 8 d-tiles of 16x16 f32 (64 VGPRs) ----
    CRegs O[8];
#pragma unroll
    for (int dt = 0; dt < 8; ++dt)
#pragma unroll
        for (int j = 0; j < 8; ++j) O[dt].f[j] = 0.0f;

    // Online-softmax row state for row (lane%16); consistently replicated in
    // both wave halves (both halves compute identical merged values).
    float mRun = -3.0e38f;
    float lRun = 0.0f;

    // cooperative-load mapping: thread -> (kv row, 32-float d-chunk)
    const int r  = tid >> 2;          // 0..63
    const int d0 = (tid & 3) << 5;    // 0,32,64,96

    for (int kvBase = 0; kvBase < S_; kvBase += BN) {
        // ============ cooperative tile load: f32 global -> bf16 LDS ============
        const float* krow = k + ((size_t)head * S_ + kvBase + r) * D_ + d0;
        const float* vrow = v + ((size_t)head * S_ + kvBase + r) * D_ + d0;
        unsigned* kd = (unsigned*)&Kld[r][d0];
        const float4* k4 = (const float4*)krow;
        const float4* v4 = (const float4*)vrow;
#pragma unroll
        for (int j = 0; j < 8; ++j) {
            float4 f = k4[j];
            kd[2 * j]     = pack2(f.x, f.y);
            kd[2 * j + 1] = pack2(f.z, f.w);
        }
#pragma unroll
        for (int j = 0; j < 8; ++j) {
            float4 f = v4[j];
            Vt[d0 + 4 * j + 0][r] = bfbits(f.x);
            Vt[d0 + 4 * j + 1][r] = bfbits(f.y);
            Vt[d0 + 4 * j + 2][r] = bfbits(f.z);
            Vt[d0 + 4 * j + 3][r] = bfbits(f.w);
        }
        if (tid < BN) ksegS[tid] = kvseg[kvBase + tid];
        __syncthreads();

        // prefetch next KV tile into L2 while we compute
        if (kvBase + BN < S_) {
            __builtin_prefetch(krow + (size_t)BN * D_, 0, 0);
            __builtin_prefetch(vrow + (size_t)BN * D_, 0, 0);
        }

        // ============ S = Q K^T (16x64 per wave, four 16x16 WMMA tiles) ============
        // B 32x16 bf16, same k order as Q's A: lane col n = lane%16; VGPR j holds K rows 16*half+2j,+1
        //  -> 16 contiguous bf16 per lane from Kld row (t*16 + lane%16).
#pragma unroll
        for (int t = 0; t < 4; ++t) {
            CRegs c;
#pragma unroll
            for (int j = 0; j < 8; ++j) c.f[j] = 0.0f;
#pragma unroll
            for (int kc = 0; kc < 4; ++kc) {
                ARegs b;
                const unsigned* src =
                    (const unsigned*)&Kld[t * 16 + laneLo][kc * 32 + 16 * half];
#pragma unroll
                for (int j = 0; j < 8; ++j) b.u[j] = src[j];
                c.v = __builtin_amdgcn_wmma_f32_16x16x32_bf16(
                          false, QA[kc].v, false, b.v, (short)0, c.v, false, false);
            }
            // mask + stage scores (C layout: row m = vv + 8*half, col n = lane%16)
            const int ks = ksegS[t * 16 + laneLo];
#pragma unroll
            for (int vv = 0; vv < 8; ++vv) {
                float s = (qs8[vv] == ks) ? c.f[vv] : -3.0e38f;
                Sld[wave][vv + 8 * half][t * 16 + laneLo] = s;
            }
        }

        // ============ online softmax: all 32 lanes ============
        // Lane handles row = lane%16, columns [32*half, 32*half+32); the two
        // half-row partials are merged with shfl_xor(16).
        {
            const float* srow = &Sld[wave][laneLo][32 * half];
            float mxp = -3.0e38f;
#pragma unroll
            for (int n = 0; n < 32; ++n) mxp = fmaxf(mxp, srow[n]);
            float mx = fmaxf(fmaxf(mxp, __shfl_xor(mxp, 16)), mRun);
            const float alpha = __expf(mRun - mx);
            float sum = 0.0f;
            unsigned* prow = (unsigned*)&Pld[wave][laneLo][0];
#pragma unroll
            for (int n = 0; n < 32; n += 2) {
                float s0 = srow[n];
                float s1 = srow[n + 1];
                float p0 = (s0 > -1.0e37f) ? __expf(s0 - mx) : 0.0f;
                float p1 = (s1 > -1.0e37f) ? __expf(s1 - mx) : 0.0f;
                sum += p0 + p1;
                prow[16 * half + (n >> 1)] = pack2(p0, p1);
            }
            sum += __shfl_xor(sum, 16);
            mRun = mx;
            lRun = lRun * alpha + sum;
            rowAlpha[wave][laneLo] = alpha;   // both halves write same value
        }

        // ============ rescale O by per-row alpha (once per 64 kv rows) ============
        float a8[8];
#pragma unroll
        for (int vv = 0; vv < 8; ++vv) a8[vv] = rowAlpha[wave][8 * half + vv];
#pragma unroll
        for (int dt = 0; dt < 8; ++dt)
#pragma unroll
            for (int vv = 0; vv < 8; ++vv) O[dt].f[vv] *= a8[vv];

        // ============ O += P V (8 d-tiles x 2 K-chunks of 32 kv) ============
        // A-layout read of P from LDS (row = lane%16); two 16x32 A tiles.
        ARegs P0, P1;
        const unsigned* ps = (const unsigned*)&Pld[wave][laneLo][0];
#pragma unroll
        for (int j = 0; j < 4; ++j) {
            P0.u[j]     = ps[8 * half + j];
            P0.u[4 + j] = ps[4 + 8 * half + j];
            P1.u[j]     = ps[16 + 8 * half + j];
            P1.u[4 + j] = ps[20 + 8 * half + j];
        }
#pragma unroll
        for (int dt = 0; dt < 8; ++dt) {
            ARegs b0, b1;  // B from V^T: row d = dt*16 + lane%16
            const unsigned* vs0 = (const unsigned*)&Vt[dt * 16 + laneLo][16 * half];
            const unsigned* vs1 = (const unsigned*)&Vt[dt * 16 + laneLo][32 + 16 * half];
#pragma unroll
            for (int j = 0; j < 8; ++j) { b0.u[j] = vs0[j]; b1.u[j] = vs1[j]; }
            O[dt].v = __builtin_amdgcn_wmma_f32_16x16x32_bf16(
                          false, P0.v, false, b0.v, (short)0, O[dt].v, false, false);
            O[dt].v = __builtin_amdgcn_wmma_f32_16x16x32_bf16(
                          false, P1.v, false, b1.v, (short)0, O[dt].v, false, false);
        }

        __syncthreads();
    }

    // ---- final 1/l normalization + store ----
    rowStat[wave][laneLo] = (lRun > 0.0f) ? (1.0f / lRun) : 0.0f;
    float li[8];
#pragma unroll
    for (int vv = 0; vv < 8; ++vv) li[vv] = rowStat[wave][8 * half + vv];

    float* orow = out + ((size_t)head * S_ + qWave) * D_;
#pragma unroll
    for (int dt = 0; dt < 8; ++dt)
#pragma unroll
        for (int vv = 0; vv < 8; ++vv)
            orow[(size_t)(8 * half + vv) * D_ + dt * 16 + laneLo] =
                O[dt].f[vv] * li[vv];
}

extern "C" void kernel_launch(void* const* d_in, const int* in_sizes, int n_in,
                              void* d_out, int out_size, void* d_ws, size_t ws_size,
                              hipStream_t stream) {
    const float* q     = (const float*)d_in[0];
    const float* k     = (const float*)d_in[1];
    const float* v     = (const float*)d_in[2];
    const int*   qseg  = (const int*)d_in[3];
    const int*   kvseg = (const int*)d_in[4];
    float*       out   = (float*)d_out;
    (void)in_sizes; (void)n_in; (void)out_size; (void)d_ws; (void)ws_size;

    dim3 grid(H_ * (S_ / BM));
    dim3 block(256);
    flash_attn_seg_kernel<<<grid, block, 0, stream>>>(q, k, v, qseg, kvseg, out);
}
